// ConcatAttention_21320217657407
// MI455X (gfx1250) — matrix operand order repaired and checked by reading, not compiled, not run on hardware
//
#include <hip/hip_runtime.h>
#include <stdint.h>

#define BS   16
#define NH   8
#define LMAX 1024
#define CC   512
#define HD   64

typedef __bf16 bf16;
typedef bf16  v16bf __attribute__((ext_vector_type(16)));
typedef float v8f   __attribute__((ext_vector_type(8)));

union FragB { v16bf v; uint4 q[2]; };

static __device__ __forceinline__ unsigned short f2bfu(float f) {
  unsigned int u = __builtin_bit_cast(unsigned int, f);
  unsigned int r = u + 0x7FFFu + ((u >> 16) & 1u);   // round-to-nearest-even
  return (unsigned short)(r >> 16);
}

static __device__ __forceinline__ v8f wmma_bf16(v16bf a, v16bf b, v8f c) {
  return __builtin_amdgcn_wmma_f32_16x16x32_bf16(false, a, false, b, (short)0, c, false, false);
}

// ---------------------------------------------------------------- kernel 0
__global__ void count_scan_kernel(const int* __restrict__ bid, int n,
                                  int* __restrict__ counts, int* __restrict__ offsets) {
  __shared__ int c[BS];
  if (threadIdx.x < BS) c[threadIdx.x] = 0;
  __syncthreads();
  for (int i = threadIdx.x; i < n; i += blockDim.x) atomicAdd(&c[bid[i]], 1);
  __syncthreads();
  if (threadIdx.x == 0) {
    int s = 0;
    for (int b = 0; b < BS; ++b) { counts[b] = c[b]; offsets[b] = s; s += c[b]; }
  }
}

// ---------------------------------------------------------------- kernel 1: fp32 weights -> bf16
__global__ void wconv_kernel(const float* __restrict__ w_in, const float* __restrict__ w_out,
                             unsigned short* __restrict__ wqkv, unsigned short* __restrict__ wo) {
  int i = blockIdx.x * blockDim.x + threadIdx.x;
  const int n1 = 3 * CC * CC;                       // 786432
  if (i < n1) wqkv[i] = f2bfu(w_in[i]);
  else        wo[i - n1] = f2bfu(w_out[i - n1]);    // total threads = n1 + CC*CC
}

// ---------------------------------------------------------------- kernel 2: ragged -> padded bf16
__global__ void pad_kernel(const float* __restrict__ feat,
                           const int* __restrict__ counts, const int* __restrict__ offsets,
                           unsigned short* __restrict__ xpad) {
  int gid = blockIdx.x * blockDim.x + threadIdx.x;  // one thread per 8 channels
  int row = gid >> 6;                               // 64 chunks per 512-wide row
  int c0  = (gid & 63) << 3;
  int b = row >> 10, l = row & 1023;
  __align__(16) unsigned short out[8];
  if (l < counts[b]) {
    int tok = offsets[b] + l;
    const float4 f0 = *reinterpret_cast<const float4*>(feat + (size_t)tok * CC + c0);
    const float4 f1 = *reinterpret_cast<const float4*>(feat + (size_t)tok * CC + c0 + 4);
    out[0]=f2bfu(f0.x); out[1]=f2bfu(f0.y); out[2]=f2bfu(f0.z); out[3]=f2bfu(f0.w);
    out[4]=f2bfu(f1.x); out[5]=f2bfu(f1.y); out[6]=f2bfu(f1.z); out[7]=f2bfu(f1.w);
  } else {
#pragma unroll
    for (int j = 0; j < 8; ++j) out[j] = 0;
  }
  *reinterpret_cast<uint4*>(xpad + (size_t)row * CC + c0) = *reinterpret_cast<uint4*>(out);
}

// ----------------------------------------------------------------
// Shared 16x64 GEMM tile core with explicit double-buffered pipeline.
// arow: A base for this lane's row (row-major, stride CC)
// wt  : B base = weights + colBase*CC  (row-major [n][k], stride CC)
static __device__ __forceinline__ void gemm16x64(
    const unsigned short* __restrict__ arow,
    const unsigned short* __restrict__ wt,
    int lm, int lh, v8f acc[4]) {
  const unsigned short* wrow[4];
#pragma unroll
  for (int t = 0; t < 4; ++t) wrow[t] = wt + (size_t)(t * 16 + lm) * CC + lh * 16;
  const unsigned short* arl = arow + lh * 16;

  FragB a0, a1, b0[4], b1[4];
  auto loadA = [&](FragB& a, int k0) {
    a.q[0] = *reinterpret_cast<const uint4*>(arl + k0);
    a.q[1] = *reinterpret_cast<const uint4*>(arl + k0 + 8);
  };
  auto loadB = [&](FragB* b, int k0) {
#pragma unroll
    for (int t = 0; t < 4; ++t) {
      b[t].q[0] = *reinterpret_cast<const uint4*>(wrow[t] + k0);
      b[t].q[1] = *reinterpret_cast<const uint4*>(wrow[t] + k0 + 8);
    }
  };
  loadA(a0, 0); loadB(b0, 0);
#pragma unroll 1
  for (int k0 = 0; k0 < CC; k0 += 64) {
    __builtin_prefetch(arl + k0 + 128, 0, 0);        // global_prefetch_b8
    loadA(a1, k0 + 32); loadB(b1, k0 + 32);          // stage k0+32
#pragma unroll
    for (int t = 0; t < 4; ++t) acc[t] = wmma_bf16(a0.v, b0[t].v, acc[t]);
    if (k0 + 64 < CC) { loadA(a0, k0 + 64); loadB(b0, k0 + 64); }  // stage k0+64
#pragma unroll
    for (int t = 0; t < 4; ++t) acc[t] = wmma_bf16(a1.v, b1[t].v, acc[t]);
  }
}

// ---------------------------------------------------------------- kernel 3: QKV projection GEMM
// grid (1536/64, 16384/64), 4 waves/block; wave computes a 16x64 tile.
__global__ void __launch_bounds__(128) qkv_kernel(
    const unsigned short* __restrict__ xpad, const unsigned short* __restrict__ wbf,
    const float* __restrict__ bias,
    unsigned short* __restrict__ Q, unsigned short* __restrict__ K,
    unsigned short* __restrict__ Vt) {
  const int lane = threadIdx.x & 31, wave = threadIdx.x >> 5;
  const int lm = lane & 15, lh = lane >> 4;
  const int rowBase = blockIdx.y * 64 + wave * 16;
  const int colBase = blockIdx.x * 64;
  v8f acc[4] = {};
  gemm16x64(xpad + (size_t)(rowBase + lm) * CC, wbf + (size_t)colBase * CC, lm, lh, acc);
#pragma unroll
  for (int t = 0; t < 4; ++t)
#pragma unroll
    for (int r = 0; r < 8; ++r) {
      int m = rowBase + r + lh * 8;
      int n = colBase + t * 16 + lm;
      float val = acc[t][r] + bias[n];
      int bb = m >> 10, ll = m & 1023;
      int sec = n >> 9, c = n & 511, h = c >> 6, d = c & 63;
      int bh = bb * NH + h;
      unsigned short bv = f2bfu(val);
      if (sec == 0)      Q[((size_t)bh * LMAX + ll) * HD + d] = bv;
      else if (sec == 1) K[((size_t)bh * LMAX + ll) * HD + d] = bv;
      else               Vt[((size_t)bh * HD + d) * LMAX + ll] = bv;  // V transposed
    }
}

// ---------------------------------------------------------------- kernel 4: flash attention
// grid (16, BS*NH), 4 waves/block; wave handles 16 query rows of one (b,h).
__global__ void __launch_bounds__(128) attn_kernel(
    const unsigned short* __restrict__ Q, const unsigned short* __restrict__ Kb,
    const unsigned short* __restrict__ Vt, const int* __restrict__ counts,
    unsigned short* __restrict__ ctx) {
  __shared__ __align__(16) unsigned short plds[4][16][32];
  const int lane = threadIdx.x & 31, wave = threadIdx.x >> 5;
  const int lm = lane & 15, lh = lane >> 4;
  const int bh = blockIdx.y;
  const int b = bh >> 3, h = bh & 7;
  const int qt = blockIdx.x * 4 + wave;
  const int count = counts[b];
  const float scale = 0.125f;                       // 1/sqrt(64)

  const unsigned short* qrow = Q + ((size_t)bh * LMAX + qt * 16 + lm) * HD;
  const unsigned short* krow0 = Kb + ((size_t)bh * LMAX + lm) * HD;       // + key*HD
  const unsigned short* krow1 = Kb + ((size_t)bh * LMAX + 16 + lm) * HD;
  FragB qa[2];
#pragma unroll
  for (int s = 0; s < 2; ++s) {
    int d0 = s * 32 + lh * 16;
    qa[s].q[0] = *reinterpret_cast<const uint4*>(qrow + d0);
    qa[s].q[1] = *reinterpret_cast<const uint4*>(qrow + d0 + 8);
  }
  float mrow[8], lrow[8];
#pragma unroll
  for (int r = 0; r < 8; ++r) { mrow[r] = -1e30f; lrow[r] = 0.f; }
  v8f o[4] = {};

  const int ktmax = (count + 31) >> 5;              // uniform over block (same b)
#pragma unroll 1
  for (int kt = 0; kt < ktmax; ++kt) {
    __builtin_prefetch(krow0 + (size_t)(kt + 1) * 32 * HD, 0, 0);
    // ---- S = scale * Q K^T  (2 halves x 2 k-steps)
    v8f s0 = {}, s1 = {};
#pragma unroll
    for (int s = 0; s < 2; ++s) {
      int d0 = s * 32 + lh * 16;
      FragB k0f, k1f;
      const unsigned short* kr0 = krow0 + (size_t)kt * 32 * HD + d0;
      k0f.q[0] = *reinterpret_cast<const uint4*>(kr0);
      k0f.q[1] = *reinterpret_cast<const uint4*>(kr0 + 8);
      s0 = wmma_bf16(qa[s].v, k0f.v, s0);
      const unsigned short* kr1 = krow1 + (size_t)kt * 32 * HD + d0;
      k1f.q[0] = *reinterpret_cast<const uint4*>(kr1);
      k1f.q[1] = *reinterpret_cast<const uint4*>(kr1 + 8);
      s1 = wmma_bf16(qa[s].v, k1f.v, s1);
    }
    // ---- V fragments issued early: independent of softmax, hides latency
    FragB vb[4];
#pragma unroll
    for (int t = 0; t < 4; ++t) {
      const unsigned short* vr =
          Vt + ((size_t)bh * HD + t * 16 + lm) * LMAX + kt * 32 + lh * 16;
      vb[t].q[0] = *reinterpret_cast<const uint4*>(vr);
      vb[t].q[1] = *reinterpret_cast<const uint4*>(vr + 8);
    }
    // ---- online softmax (row stats via xor-shuffles within 16-lane halves)
    const bool msk0 = (kt * 32 + lm) >= count;
    const bool msk1 = (kt * 32 + 16 + lm) >= count;
    float alpha[8];
#pragma unroll
    for (int r = 0; r < 8; ++r) {
      float a0 = msk0 ? -1e30f : s0[r] * scale;
      float a1 = msk1 ? -1e30f : s1[r] * scale;
      float mt = fmaxf(a0, a1);
      mt = fmaxf(mt, __shfl_xor(mt, 1));
      mt = fmaxf(mt, __shfl_xor(mt, 2));
      mt = fmaxf(mt, __shfl_xor(mt, 4));
      mt = fmaxf(mt, __shfl_xor(mt, 8));
      float mn = fmaxf(mrow[r], mt);
      float al = __expf(mrow[r] - mn);
      mrow[r] = mn;
      float p0 = __expf(a0 - mn);
      float p1 = __expf(a1 - mn);
      s0[r] = p0; s1[r] = p1;
      float rs = p0 + p1;
      rs += __shfl_xor(rs, 1);
      rs += __shfl_xor(rs, 2);
      rs += __shfl_xor(rs, 4);
      rs += __shfl_xor(rs, 8);
      lrow[r] = lrow[r] * al + rs;
      alpha[r] = al;
    }
#pragma unroll
    for (int t = 0; t < 4; ++t)
#pragma unroll
      for (int r = 0; r < 8; ++r) o[t][r] *= alpha[r];
    // ---- P (C-layout f32) -> bf16 A-layout via LDS bounce
#pragma unroll
    for (int r = 0; r < 8; ++r) {
      plds[wave][lh * 8 + r][lm]      = f2bfu(s0[r]);
      plds[wave][lh * 8 + r][lm + 16] = f2bfu(s1[r]);
    }
    __syncthreads();
    FragB pa;
    pa.q[0] = *reinterpret_cast<const uint4*>(&plds[wave][lm][lh * 16]);
    pa.q[1] = *reinterpret_cast<const uint4*>(&plds[wave][lm][lh * 16 + 8]);
    __syncthreads();
    // ---- O += P V
#pragma unroll
    for (int t = 0; t < 4; ++t) o[t] = wmma_bf16(pa.v, vb[t].v, o[t]);
  }
  float inv[8];
#pragma unroll
  for (int r = 0; r < 8; ++r) inv[r] = 1.0f / fmaxf(lrow[r], 1e-30f);
#pragma unroll
  for (int t = 0; t < 4; ++t)
#pragma unroll
    for (int r = 0; r < 8; ++r) {
      int lq = qt * 16 + lh * 8 + r;
      int d  = t * 16 + lm;
      ctx[((size_t)b * LMAX + lq) * CC + h * HD + d] = f2bfu(o[t][r] * inv[r]);
    }
}

// ---------------------------------------------------------------- kernel 5: out-proj GEMM + gather
__global__ void __launch_bounds__(128) oproj_kernel(
    const unsigned short* __restrict__ ctx, const unsigned short* __restrict__ wo,
    const float* __restrict__ bias, const int* __restrict__ counts,
    const int* __restrict__ offsets, float* __restrict__ out) {
  const int lane = threadIdx.x & 31, wave = threadIdx.x >> 5;
  const int lm = lane & 15, lh = lane >> 4;
  const int rowBase = blockIdx.y * 64 + wave * 16;
  const int colBase = blockIdx.x * 64;
  v8f acc[4] = {};
  gemm16x64(ctx + (size_t)(rowBase + lm) * CC, wo + (size_t)colBase * CC, lm, lh, acc);
#pragma unroll
  for (int t = 0; t < 4; ++t)
#pragma unroll
    for (int r = 0; r < 8; ++r) {
      int m = rowBase + r + lh * 8;
      int n = colBase + t * 16 + lm;
      int bb = m >> 10, ll = m & 1023;
      if (ll < counts[bb]) {
        int tok = offsets[bb] + ll;
        out[(size_t)tok * CC + n] = acc[t][r] + bias[n];
      }
    }
}

// ---------------------------------------------------------------- launch
extern "C" void kernel_launch(void* const* d_in, const int* in_sizes, int n_in,
                              void* d_out, int out_size, void* d_ws, size_t ws_size,
                              hipStream_t stream) {
  const int*   batch_ids = (const int*)d_in[0];
  const float* features  = (const float*)d_in[1];
  const float* in_w      = (const float*)d_in[2];
  const float* in_b      = (const float*)d_in[3];
  const float* out_w     = (const float*)d_in[4];
  const float* out_b     = (const float*)d_in[5];
  const int n = in_sizes[0];

  char* w = (char*)d_ws;
  int* counts  = (int*)w;
  int* offsets = counts + BS;
  size_t off = 256;
  const size_t padElems = (size_t)BS * LMAX * CC;   // 8,388,608
  unsigned short* xpad = (unsigned short*)(w + off); off += padElems * 2;
  unsigned short* Q    = (unsigned short*)(w + off); off += padElems * 2;
  unsigned short* K    = (unsigned short*)(w + off); off += padElems * 2;
  unsigned short* Vt   = (unsigned short*)(w + off); off += padElems * 2;
  unsigned short* ctx  = (unsigned short*)(w + off); off += padElems * 2;
  unsigned short* wqkv = (unsigned short*)(w + off); off += (size_t)3 * CC * CC * 2;
  unsigned short* wo   = (unsigned short*)(w + off); off += (size_t)CC * CC * 2;
  (void)ws_size; (void)n_in; (void)out_size;

  count_scan_kernel<<<1, 256, 0, stream>>>(batch_ids, n, counts, offsets);
  wconv_kernel<<<4096, 256, 0, stream>>>(in_w, out_w, wqkv, wo);       // 1,048,576 threads
  pad_kernel<<<4096, 256, 0, stream>>>(features, counts, offsets, xpad);
  qkv_kernel<<<dim3(24, 256), 128, 0, stream>>>(xpad, wqkv, in_b, Q, K, Vt);
  attn_kernel<<<dim3(16, BS * NH), 128, 0, stream>>>(Q, K, Vt, counts, ctx);
  oproj_kernel<<<dim3(8, 256), 128, 0, stream>>>(ctx, wo, out_b, counts, offsets,
                                                 (float*)d_out);
}
